// GraphAttentionLayer_42631845380540
// MI455X (gfx1250) — matrix operand order repaired and checked by reading, not compiled, not run on hardware
//
#include <hip/hip_runtime.h>

typedef __attribute__((ext_vector_type(16))) _Float16 v16h;
typedef __attribute__((ext_vector_type(8)))  float    v8f;

#define GAT_N     8192
#define GAT_HEADS 2
#define GAT_INF   256
#define GAT_OUTF  64
#define RBLK      16
#define JBLK      128
#define NSPLIT    4
#define JITER     (GAT_N / JBLK / NSPLIT)   // 16
#define NEG_INF   (-__builtin_inff())

// ---------------------------------------------------------------------------
// Kernel 1: h = x @ W (f32), emit h16T[head][o][node] (f16, transposed) and
// s[head][node] = h . a_src, d[head][node] = h . a_dst  (f32)
// grid: 512 blocks x 128 threads; each block = 16 nodes
// ---------------------------------------------------------------------------
__global__ __launch_bounds__(128) void k_proj(
    const float* __restrict__ x, const float* __restrict__ W,
    const float* __restrict__ a_src, const float* __restrict__ a_dst,
    _Float16* __restrict__ h16T, float* __restrict__ sArr, float* __restrict__ dArr)
{
    __shared__ float    xs[RBLK][GAT_INF];     // 16 KB
    __shared__ float    hf[RBLK][128];         // 8 KB
    __shared__ _Float16 h16s[RBLK][128];       // 4 KB

    const int nb  = blockIdx.x;
    const int tid = threadIdx.x;

    // load 16 node rows of x (4096 floats) as float4
    const float4* xsrc = (const float4*)(x + (size_t)nb * RBLK * GAT_INF);
    float4*       xdst = (float4*)(&xs[0][0]);
#pragma unroll
    for (int it = 0; it < 8; ++it) xdst[tid + 128 * it] = xsrc[tid + 128 * it];
    __syncthreads();

    const int head = tid >> 6, o = tid & 63;
    const float* wcol = W + head * GAT_INF * GAT_OUTF + o;   // stride 64 per i
    for (int nl = 0; nl < RBLK; ++nl) {
        float acc = 0.f;
#pragma unroll 8
        for (int i = 0; i < GAT_INF; ++i) acc = fmaf(xs[nl][i], wcol[i * GAT_OUTF], acc);
        hf[nl][tid]   = acc;
        h16s[nl][tid] = (_Float16)acc;
    }
    __syncthreads();

    if (tid < 32) {                       // s, d reductions (f32 h)
        const int hh = tid >> 4, row = tid & 15;
        float s = 0.f, d = 0.f;
#pragma unroll 8
        for (int oo = 0; oo < GAT_OUTF; ++oo) {
            const float hv = hf[row][hh * GAT_OUTF + oo];
            s = fmaf(hv, a_src[hh * GAT_OUTF + oo], s);
            d = fmaf(hv, a_dst[hh * GAT_OUTF + oo], d);
        }
        sArr[hh * GAT_N + nb * RBLK + row] = s;
        dArr[hh * GAT_N + nb * RBLK + row] = d;
    }

    // transposed f16 store: h16T[head][o][node], 16 contiguous nodes per thread
    union { _Float16 h[16]; uint4 u[2]; } tmp;
#pragma unroll
    for (int nl = 0; nl < RBLK; ++nl) tmp.h[nl] = h16s[nl][tid];
    uint4* dst = (uint4*)(h16T + ((size_t)(head * GAT_OUTF + o)) * GAT_N + (size_t)nb * RBLK);
    dst[0] = tmp.u[0];
    dst[1] = tmp.u[1];
}

// ---------------------------------------------------------------------------
// Kernel 2: flash-attention style online softmax + P@H via WMMA f16->f32.
// grid: 512 row-blocks * 4 K-splits = 2048 blocks x 128 threads (4 waves).
// wave w: head = w>>1, column half = (w&1)*32 of the 16x64 accumulator.
// Both heads share each adjacency tile (single HBM pass over adj).
// ---------------------------------------------------------------------------
__global__ __launch_bounds__(128) void k_attn(
    const float* __restrict__ adj, const _Float16* __restrict__ h16T,
    const float* __restrict__ sArr, const float* __restrict__ dArr,
    float* __restrict__ partM, float* __restrict__ partL, float* __restrict__ partAcc)
{
    const int rb  = blockIdx.x >> 2;
    const int sp  = blockIdx.x & 3;
    const int tid = threadIdx.x;
    const int lane = tid & 31;
    const int wv   = tid >> 5;

    __shared__ float    adjT[RBLK][132];          // padded pitch
    __shared__ _Float16 pT[GAT_HEADS][RBLK][132]; // probs, WMMA-A source
    __shared__ _Float16 hTs[GAT_HEADS][GAT_OUTF][132]; // h tile, n-major (K contiguous)
    __shared__ float    sRow[GAT_HEADS][RBLK];
    __shared__ float    dCol[GAT_HEADS][JBLK];
    __shared__ float    mArr[GAT_HEADS][RBLK], lArr[GAT_HEADS][RBLK], scArr[GAT_HEADS][RBLK];
    __shared__ float    part[GAT_HEADS][RBLK][4];

    if (tid < 32) {
        const int hh = tid >> 4, r = tid & 15;
        sRow[hh][r] = sArr[hh * GAT_N + rb * RBLK + r];
        mArr[hh][r] = NEG_INF;
        lArr[hh][r] = 0.f;
    }

    const int hhw   = wv >> 1;            // wave's head
    const int nbase = (wv & 1) * 32;      // wave's column half
    const int m_a   = lane & 15;          // A-row / B-col within 16-tile
    const int roff  = (lane < 16) ? 0 : 8;
    v8f acc0 = {}; v8f acc1 = {};

    for (int jl = 0; jl < JITER; ++jl) {
        const int jb = sp * JITER + jl;   // global 128-column block
        __syncthreads();                  // protect LDS reuse across iterations

        // ---- phase 1: stage adj tile, d-columns, h tile ----
        {
            const int r = tid >> 3, c0 = (tid & 7) * 16;
            const float4* src = (const float4*)(adj + (size_t)(rb * RBLK + r) * GAT_N + jb * JBLK + c0);
#pragma unroll
            for (int q = 0; q < 4; ++q) *(float4*)&adjT[r][c0 + q * 4] = src[q];
        }
        dCol[tid >> 7][tid & 127]         = dArr[(tid >> 7) * GAT_N + jb * JBLK + (tid & 127)];
        dCol[(tid + 128) >> 7][tid & 127] = dArr[((tid + 128) >> 7) * GAT_N + jb * JBLK + (tid & 127)];
        {
            const int hh = tid >> 6, n = tid & 63;
            const uint2* src = (const uint2*)(h16T + ((size_t)hh * GAT_OUTF + n) * GAT_N + (size_t)jb * JBLK);
            uint2* dst = (uint2*)&hTs[hh][n][0];
#pragma unroll 8
            for (int q = 0; q < 32; ++q) dst[q] = src[q];
        }
        __syncthreads();

        // ---- phase 2: e = leakyrelu(s_i + d_j), mask, online softmax ----
        const int hh2  = tid >> 6;
        const int row2 = (tid >> 2) & 15;
        const int sub  = tid & 3;
        const int gi   = rb * RBLK + row2;
        const int jcb  = jb * JBLK + sub * 32;
        const float sv = sRow[hh2][row2];
        float ev[32];
        float tmax = NEG_INF;
#pragma unroll
        for (int c = 0; c < 32; ++c) {
            const float av = adjT[row2][sub * 32 + c];
            float e = sv + dCol[hh2][sub * 32 + c];
            e = (e > 0.f) ? e : 0.2f * e;
            const bool keep = (av != 0.f) || (gi == (jcb + c));
            e = keep ? e : NEG_INF;
            ev[c] = e;
            tmax  = fmaxf(tmax, e);
        }
        part[hh2][row2][sub] = tmax;
        __syncthreads();
        if (sub == 0) {
            const float t  = fmaxf(fmaxf(part[hh2][row2][0], part[hh2][row2][1]),
                                   fmaxf(part[hh2][row2][2], part[hh2][row2][3]));
            const float mo = mArr[hh2][row2];
            const float mn = fmaxf(mo, t);
            scArr[hh2][row2] = (mn > NEG_INF) ? __expf(mo - mn) : 1.0f;
            mArr[hh2][row2]  = mn;
        }
        __syncthreads();
        const float mn = mArr[hh2][row2];
        float psum = 0.f;
#pragma unroll
        for (int c = 0; c < 32; ++c) {
            float p = 0.f;
            if (ev[c] > NEG_INF) p = __expf(ev[c] - mn);
            psum += p;
            pT[hh2][row2][sub * 32 + c] = (_Float16)p;
        }
        part[hh2][row2][sub] = psum;
        __syncthreads();
        if (sub == 0) {
            lArr[hh2][row2] = lArr[hh2][row2] * scArr[hh2][row2]
                + part[hh2][row2][0] + part[hh2][row2][1]
                + part[hh2][row2][2] + part[hh2][row2][3];
        }

        // ---- phase 3: rescale accumulators, P @ H via WMMA ----
        float sc[8];
#pragma unroll
        for (int r = 0; r < 8; ++r) sc[r] = scArr[hhw][r + roff];
#pragma unroll
        for (int r = 0; r < 8; ++r) { acc0[r] *= sc[r]; acc1[r] *= sc[r]; }

#pragma unroll
        for (int kk = 0; kk < 4; ++kk) {
            v16h a, b0, b1;
            // A fragment: 16-bit A 16x32 layout (ISA 7.12.2)
#pragma unroll
            for (int v = 0; v < 8; ++v) {
                const int kp = kk * 32 + ((v < 4) ? 2 * v : 16 + 2 * (v - 4)) + ((lane < 16) ? 0 : 8);
                a[2 * v]     = pT[hhw][m_a][kp];
                a[2 * v + 1] = pT[hhw][m_a][kp + 1];
            }
            // B fragments: same K layout as A (16-bit 16x32 table order)
#pragma unroll
            for (int v = 0; v < 8; ++v) {
                const int kq = kk * 32 + ((v < 4) ? 2 * v : 16 + 2 * (v - 4)) + ((lane < 16) ? 0 : 8);
                b0[2 * v]     = hTs[hhw][nbase + m_a][kq];
                b0[2 * v + 1] = hTs[hhw][nbase + m_a][kq + 1];
                b1[2 * v]     = hTs[hhw][nbase + 16 + m_a][kq];
                b1[2 * v + 1] = hTs[hhw][nbase + 16 + m_a][kq + 1];
            }
            acc0 = __builtin_amdgcn_wmma_f32_16x16x32_f16(false, a, false, b0, (short)0, acc0, false, false);
            acc1 = __builtin_amdgcn_wmma_f32_16x16x32_f16(false, a, false, b1, (short)0, acc1, false, false);
        }
    }
    __syncthreads();

    // ---- write per-split partial state ----
    const size_t pbase = ((size_t)rb * NSPLIT + sp) * GAT_HEADS;
    if (tid < 32) {
        const int hh = tid >> 4, r = tid & 15;
        partM[(pbase + hh) * RBLK + r] = mArr[hh][r];
        partL[(pbase + hh) * RBLK + r] = lArr[hh][r];
    }
    float* pa = partAcc + (pbase + hhw) * (RBLK * GAT_OUTF);
#pragma unroll
    for (int r = 0; r < 8; ++r) {
        pa[(r + roff) * GAT_OUTF + nbase + m_a]      = acc0[r];
        pa[(r + roff) * GAT_OUTF + nbase + 16 + m_a] = acc1[r];
    }
}

// ---------------------------------------------------------------------------
// Kernel 3: combine the NSPLIT partial softmax states, normalize, add bias.
// grid: 8192*128/256 = 4096 blocks x 256 threads
// ---------------------------------------------------------------------------
__global__ __launch_bounds__(256) void k_combine(
    const float* __restrict__ partM, const float* __restrict__ partL,
    const float* __restrict__ partAcc, const float* __restrict__ bias,
    float* __restrict__ out)
{
    const int idx  = blockIdx.x * 256 + threadIdx.x;
    const int node = idx >> 7;
    const int c    = idx & 127;
    const int hh   = c >> 6, o = c & 63;
    const int rb   = node >> 4, r = node & 15;

    float mv[NSPLIT];
    float mstar = NEG_INF;
#pragma unroll
    for (int s = 0; s < NSPLIT; ++s) {
        mv[s] = partM[((size_t)(rb * NSPLIT + s) * GAT_HEADS + hh) * RBLK + r];
        mstar = fmaxf(mstar, mv[s]);
    }
    float num = 0.f, den = 0.f;
#pragma unroll
    for (int s = 0; s < NSPLIT; ++s) {
        const size_t b = (size_t)(rb * NSPLIT + s) * GAT_HEADS + hh;
        const float w = (mv[s] > NEG_INF) ? __expf(mv[s] - mstar) : 0.f;
        num += w * partAcc[b * (RBLK * GAT_OUTF) + r * GAT_OUTF + o];
        den += w * partL[b * RBLK + r];
    }
    out[idx] = num / den + bias[c];
}

// ---------------------------------------------------------------------------
extern "C" void kernel_launch(void* const* d_in, const int* in_sizes, int n_in,
                              void* d_out, int out_size, void* d_ws, size_t ws_size,
                              hipStream_t stream)
{
    const float* x     = (const float*)d_in[0];
    const float* W     = (const float*)d_in[1];
    const float* a_src = (const float*)d_in[2];
    const float* a_dst = (const float*)d_in[3];
    const float* bias  = (const float*)d_in[4];
    const float* adj   = (const float*)d_in[5];
    float*       out   = (float*)d_out;

    char* ws = (char*)d_ws;
    _Float16* h16T = (_Float16*)(ws);                         // 2*64*8192*2  = 2 MB
    float* sArr    = (float*)(ws + 2097152);                  // 64 KB
    float* dArr    = (float*)(ws + 2162688);                  // 64 KB
    float* partM   = (float*)(ws + 2228224);                  // 256 KB
    float* partL   = (float*)(ws + 2490368);                  // 256 KB
    float* partAcc = (float*)(ws + 2752512);                  // 16 MB

    k_proj<<<GAT_N / RBLK, 128, 0, stream>>>(x, W, a_src, a_dst, h16T, sArr, dArr);
    k_attn<<<(GAT_N / RBLK) * NSPLIT, 128, 0, stream>>>(adj, h16T, sArr, dArr,
                                                        partM, partL, partAcc);
    k_combine<<<(GAT_N * GAT_HEADS * GAT_OUTF) / 256, 256, 0, stream>>>(
        partM, partL, partAcc, bias, out);
}
